// Net_83794811945439
// MI455X (gfx1250) — matrix operand order repaired and checked by reading, not compiled, not run on hardware
//
#include <hip/hip_runtime.h>

typedef _Float16 v16h __attribute__((ext_vector_type(16)));
typedef _Float16 h2   __attribute__((ext_vector_type(2)));
typedef float    v8f  __attribute__((ext_vector_type(8)));
typedef unsigned int u32x4 __attribute__((ext_vector_type(4)));
typedef int          i32x8 __attribute__((ext_vector_type(8)));
typedef int          i32x4 __attribute__((ext_vector_type(4)));

#define B_TOTAL 65536
#define D_IN    784
#define H_DIM   100
#define HP      112      // H padded to 7 tiles of 16 (channel 111 = constant-spike bias channel)
#define O_DIM   2
#define T_STEPS 25
#define KC1     25       // ceil(784/32) K-chunks for GEMM1
#define KC2     4        // K = 128 (H padded) -> 4 chunks of 32
#define NT_H    7        // 7 output tiles of 16 for H
#define BETA    0.9f
#define THRESH  1.0f

#define NFRAG_W1 (KC1*NT_H)                     // 175
#define NFRAG_W2 (KC2*NT_H)                     // 28
#define NFRAG_W3 (KC2)                          // 4
#define NFRAG    (NFRAG_W1+NFRAG_W2+NFRAG_W3)   // 207
#define FRAG_HALVES 512                          // 32 lanes * 16 halves
#define B1P_OFF_BYTES ((size_t)NFRAG*FRAG_HALVES*2)

// ---------------------------------------------------------------------------
// Prep kernel: pack w1/w2/w3 (+ folded biases) into WMMA A-fragment layout.
// A k order used here (contiguous; B matches): lane l (row M = l&15), element e:
//   K = e + (l<16 ? 0 : 16)
// ---------------------------------------------------------------------------
__global__ void pack_weights(const float* __restrict__ w1, const float* __restrict__ b1,
                             const float* __restrict__ w2, const float* __restrict__ b2,
                             const float* __restrict__ w3, const float* __restrict__ b3,
                             _Float16* __restrict__ packW, float* __restrict__ b1p)
{
    const int f    = blockIdx.x;       // fragment id
    const int lane = threadIdx.x;      // 0..31
    const int l15  = lane & 15;
    const int ksh  = (lane < 16) ? 0 : 16;

    _Float16 vals[16];
    if (f < NFRAG_W1) {                          // ---- w1 fragments
        const int kc = f / NT_H, mt = f % NT_H;
        const int M  = mt*16 + l15;
#pragma unroll
        for (int e = 0; e < 16; ++e) {
            const int K = kc*32 + e + ksh;
            float v = 0.0f;
            if (M < H_DIM && K < D_IN) v = w1[M*D_IN + K];
            vals[e] = (_Float16)v;
        }
    } else if (f < NFRAG_W1 + NFRAG_W2) {        // ---- w2 fragments (+b2 in col 111)
        const int f2 = f - NFRAG_W1;
        const int kc = f2 / NT_H, mt = f2 % NT_H;
        const int M  = mt*16 + l15;
#pragma unroll
        for (int e = 0; e < 16; ++e) {
            const int K = kc*32 + e + ksh;
            float v = 0.0f;
            if (M < H_DIM) {
                if (K < H_DIM)       v = w2[M*H_DIM + K];
                else if (K == HP-1)  v = b2[M];          // bias via constant-spike channel
            } else if (M == HP-1 && K == HP-1) {
                v = 2.0f;                                 // keep channel 111 spiking in layer 2
            }
            vals[e] = (_Float16)v;
        }
    } else {                                      // ---- w3 fragments (+b3 in col 111)
        const int kc = f - NFRAG_W1 - NFRAG_W2;
        const int M  = l15;
#pragma unroll
        for (int e = 0; e < 16; ++e) {
            const int K = kc*32 + e + ksh;
            float v = 0.0f;
            if (M < O_DIM) {
                if (K < H_DIM)       v = w3[M*H_DIM + K];
                else if (K == HP-1)  v = b3[M];
            }
            vals[e] = (_Float16)v;
        }
    }

    _Float16* dst = packW + (size_t)f*FRAG_HALVES + lane*16;
#pragma unroll
    for (int e = 0; e < 16; ++e) dst[e] = vals[e];

    if (f == 0) {   // b1 padded; b1p[111]=2.0 drives the always-spiking bias channel
        for (int i = lane; i < HP; i += 32)
            b1p[i] = (i < H_DIM) ? b1[i] : ((i == HP-1) ? 2.0f : 0.0f);
    }
}

// ---------------------------------------------------------------------------
// Build next-layer B fragment (K x 16batch) from spike D-layout using a
// half-wave SWAPX16 ds_swizzle. pkE/pkO = packed f16 spike pairs of tiles
// 2*kc and 2*kc+1 (4 dwords each, element j pairs).
// B k order used (same as A packing): lanes 0-15 hold K=0..15, lanes 16-31 hold K=16..31.
// ---------------------------------------------------------------------------
__device__ __forceinline__ v16h build_bfrag(const unsigned pkE[4], const unsigned pkO[4],
                                            int half)
{
    v16h b;
#pragma unroll
    for (int i = 0; i < 4; ++i) {
        const unsigned own  = half ? pkO[i] : pkE[i];
        const unsigned send = half ? pkE[i] : pkO[i];
        const unsigned rec  = (unsigned)__builtin_amdgcn_ds_swizzle((int)send, 0x401F); // SWAPX16
        const unsigned lo   = half ? rec : own;
        const unsigned hi   = half ? own : rec;
        const h2 l = __builtin_bit_cast(h2, lo);
        const h2 h = __builtin_bit_cast(h2, hi);
        b[2*i]     = l[0];  b[2*i+1]   = l[1];
        b[8 + 2*i] = h[0];  b[8 + 2*i+1] = h[1];
    }
    return b;
}

#define WMMA(A, Bf, C) \
    __builtin_amdgcn_wmma_f32_16x16x32_f16(false, (A), false, (Bf), (short)0, (C), false, false)

// ---------------------------------------------------------------------------
// Fused SNN kernel: each wave owns 16 batch rows; all state in VGPRs.
// ---------------------------------------------------------------------------
__global__ __launch_bounds__(256) void snn_fused(const float* __restrict__ x,
                                                 const _Float16* __restrict__ packW,
                                                 const float* __restrict__ b1p,
                                                 float* __restrict__ out)
{
    __shared__ alignas(32) _Float16 wsm[(NFRAG_W2 + NFRAG_W3) * FRAG_HALVES]; // 32 KB

    // ---- Stage w2/w3 fragments into LDS once per block via the Tensor Data
    // Mover: 2D descriptor, 4096 x 8-byte elements, contiguous (32 KB).
    {
        const unsigned long long ga =
            (unsigned long long)(const void*)(packW + (size_t)NFRAG_W1 * FRAG_HALVES);
        const unsigned ldsoff =
            (unsigned)(unsigned long long)(__attribute__((address_space(3))) const _Float16*)wsm;

        u32x4 g0;
        g0[0] = 1u;                                    // count=1 (valid user descriptor)
        g0[1] = ldsoff;                                // lds_addr (bytes)
        g0[2] = (unsigned)ga;                          // global_addr[31:0]
        g0[3] = (unsigned)((ga >> 32) & 0x01FFFFFFull) // global_addr[56:32]
              | (2u << 30);                            // type=2 ("image")

        i32x8 g1;
        g1[0] = (int)(3u << 16);                       // data_size=3 (8 bytes)
        g1[1] = (int)(0x1000u << 16);                  // tensor_dim0[15:0]=4096 (bits 63:48)
        g1[2] = (int)(1u << 16);                       // tensor_dim1[15:0]=1   (bits 95:80)
        g1[3] = (int)(0x1000u << 16);                  // tile_dim0=4096        (bits 127:112)
        g1[4] = 1;                                     // tile_dim1=1, tile_dim2=0
        g1[5] = 0x1000;                                // tensor_dim0_stride=4096
        g1[6] = 0;
        g1[7] = 0;
        const i32x4 z4 = {0, 0, 0, 0};                 // groups 2/3 unused (2D)
        const i32x8 z8 = {0, 0, 0, 0, 0, 0, 0, 0};     // extra group (clang-23 6-arg form)

        if (threadIdx.x < 32) {                        // wave 0 issues the DMA
            __builtin_amdgcn_tensor_load_to_lds(g0, g1, z4, z4, z8, 0);
            __builtin_amdgcn_s_wait_tensorcnt(0);
        }
    }
    __syncthreads();

    const int lane   = threadIdx.x & 31;
    const int wave   = threadIdx.x >> 5;
    const int l15    = lane & 15;
    const int half   = lane >> 4;
    const int batch0 = (blockIdx.x * 8 + wave) * 16;
    const int bidx   = batch0 + l15;

    // ---------------- GEMM1: cur1^T = w1 @ x^T (f16 WMMA, f32 accum) -------
    v8f cur1[NT_H];
#pragma unroll
    for (int mt = 0; mt < NT_H; ++mt)
#pragma unroll
        for (int j = 0; j < 8; ++j) cur1[mt][j] = 0.0f;

    const float* xrow = x + (size_t)bidx * D_IN;
    for (int kc = 0; kc < KC1; ++kc) {
        v16h bf;
        const int kbase = kc*32 + half*16;
        if (kbase < D_IN) {
            const float4* p = (const float4*)(xrow + kbase);
            const float4 q0 = p[0], q1 = p[1], q2 = p[2], q3 = p[3];
            __builtin_prefetch(xrow + kbase + 128, 0, 1);   // pull next chunks' lines
            bf[0]=(_Float16)q0.x; bf[1]=(_Float16)q0.y; bf[2]=(_Float16)q0.z; bf[3]=(_Float16)q0.w;
            bf[4]=(_Float16)q1.x; bf[5]=(_Float16)q1.y; bf[6]=(_Float16)q1.z; bf[7]=(_Float16)q1.w;
            bf[8]=(_Float16)q2.x; bf[9]=(_Float16)q2.y; bf[10]=(_Float16)q2.z; bf[11]=(_Float16)q2.w;
            bf[12]=(_Float16)q3.x; bf[13]=(_Float16)q3.y; bf[14]=(_Float16)q3.z; bf[15]=(_Float16)q3.w;
        } else {
#pragma unroll
            for (int e = 0; e < 16; ++e) bf[e] = (_Float16)0.0f;
        }
#pragma unroll
        for (int mt = 0; mt < NT_H; ++mt) {
            const v16h a = *(const v16h*)(packW + (size_t)(kc*NT_H + mt)*FRAG_HALVES + lane*16);
            cur1[mt] = WMMA(a, bf, cur1[mt]);
        }
    }
    // add padded b1 once (cur1 is loop-invariant); D-layout row M = j + 8*half
#pragma unroll
    for (int mt = 0; mt < NT_H; ++mt)
#pragma unroll
        for (int j = 0; j < 8; ++j)
            cur1[mt][j] += b1p[mt*16 + half*8 + j];

    // ---------------- LIF state ------------------------------------------
    v8f m1[NT_H], m2[NT_H], m3;
#pragma unroll
    for (int mt = 0; mt < NT_H; ++mt)
#pragma unroll
        for (int j = 0; j < 8; ++j) { m1[mt][j] = 0.0f; m2[mt][j] = 0.0f; }
#pragma unroll
    for (int j = 0; j < 8; ++j) m3[j] = 0.0f;

    float* spk_out = out;
    float* mem_out = out + (size_t)T_STEPS * B_TOTAL * O_DIM;

#pragma unroll 1
    for (int t = 0; t < T_STEPS; ++t) {
        // ---- layer 1 LIF + f16 spike pack (D layout) ----
        unsigned pk1[NT_H + 1][4];
#pragma unroll
        for (int i = 0; i < 4; ++i) pk1[NT_H][i] = 0u;   // K-pad tile (h 112..127)
#pragma unroll
        for (int mt = 0; mt < NT_H; ++mt) {
#pragma unroll
            for (int i = 0; i < 4; ++i) {
                h2 sp;
#pragma unroll
                for (int jj = 0; jj < 2; ++jj) {
                    const int j = 2*i + jj;
                    float m = m1[mt][j];
                    const float rst = (m > THRESH) ? 1.0f : 0.0f;
                    m = BETA*m + cur1[mt][j] - rst;
                    m1[mt][j] = m;
                    sp[jj] = (m > THRESH) ? (_Float16)1.0f : (_Float16)0.0f;
                }
                pk1[mt][i] = __builtin_bit_cast(unsigned, sp);
            }
        }

        // ---- layer 2: fold leak+reset into accumulator, WMMA-accumulate cur2 ----
#pragma unroll
        for (int mt = 0; mt < NT_H; ++mt)
#pragma unroll
            for (int j = 0; j < 8; ++j) {
                const float m = m2[mt][j];
                m2[mt][j] = BETA*m - ((m > THRESH) ? 1.0f : 0.0f);
            }
#pragma unroll
        for (int kc = 0; kc < KC2; ++kc) {
            const v16h bf = build_bfrag(pk1[2*kc], pk1[2*kc + 1], half);
#pragma unroll
            for (int mt = 0; mt < NT_H; ++mt) {
                const v16h a = *(const v16h*)&wsm[(kc*NT_H + mt)*FRAG_HALVES + lane*16];
                m2[mt] = WMMA(a, bf, m2[mt]);
            }
        }

        // ---- layer 2 spikes ----
        unsigned pk2[NT_H + 1][4];
#pragma unroll
        for (int i = 0; i < 4; ++i) pk2[NT_H][i] = 0u;
#pragma unroll
        for (int mt = 0; mt < NT_H; ++mt) {
#pragma unroll
            for (int i = 0; i < 4; ++i) {
                h2 sp;
#pragma unroll
                for (int jj = 0; jj < 2; ++jj) {
                    const float m = m2[mt][2*i + jj];
                    sp[jj] = (m > THRESH) ? (_Float16)1.0f : (_Float16)0.0f;
                }
                pk2[mt][i] = __builtin_bit_cast(unsigned, sp);
            }
        }

        // ---- layer 3 ----
#pragma unroll
        for (int j = 0; j < 8; ++j) {
            const float m = m3[j];
            m3[j] = BETA*m - ((m > THRESH) ? 1.0f : 0.0f);
        }
#pragma unroll
        for (int kc = 0; kc < KC2; ++kc) {
            const v16h bf = build_bfrag(pk2[2*kc], pk2[2*kc + 1], half);
            const v16h a  = *(const v16h*)&wsm[(NFRAG_W2 + kc)*FRAG_HALVES + lane*16];
            m3 = WMMA(a, bf, m3);
        }

        // ---- record spk3 / mem3 (valid outputs live in half 0, elements 0,1) ----
        if (half == 0) {
            const float s0 = (m3[0] > THRESH) ? 1.0f : 0.0f;
            const float s1 = (m3[1] > THRESH) ? 1.0f : 0.0f;
            const size_t o = ((size_t)t * B_TOTAL + bidx) * O_DIM;
            float2 sv; sv.x = s0;    sv.y = s1;
            float2 mv; mv.x = m3[0]; mv.y = m3[1];
            *(float2*)(spk_out + o) = sv;
            *(float2*)(mem_out + o) = mv;
        }
    }
}

extern "C" void kernel_launch(void* const* d_in, const int* in_sizes, int n_in,
                              void* d_out, int out_size, void* d_ws, size_t ws_size,
                              hipStream_t stream)
{
    (void)in_sizes; (void)n_in; (void)out_size; (void)ws_size;
    const float* x  = (const float*)d_in[0];
    const float* w1 = (const float*)d_in[1];
    const float* b1 = (const float*)d_in[2];
    const float* w2 = (const float*)d_in[3];
    const float* b2 = (const float*)d_in[4];
    const float* w3 = (const float*)d_in[5];
    const float* b3 = (const float*)d_in[6];

    _Float16* packW = (_Float16*)d_ws;
    float*    b1p   = (float*)((char*)d_ws + B1P_OFF_BYTES);
    float*    out   = (float*)d_out;

    hipLaunchKernelGGL(pack_weights, dim3(NFRAG), dim3(32), 0, stream,
                       w1, b1, w2, b2, w3, b3, packW, b1p);
    hipLaunchKernelGGL(snn_fused, dim3(B_TOTAL / 128), dim3(256), 0, stream,
                       x, packW, b1p, out);
}
